// STGC_13889924235780
// MI455X (gfx1250) — matrix operand order repaired and checked by reading, not compiled, not run on hardware
//
#include <hip/hip_runtime.h>
#include <hip/hip_bf16.h>

// ---------------------------------------------------------------------------
// STGC fused forward for MI455X (gfx1250, wave32, WMMA bf16).
// One workgroup (256 threads = 8 waves) per batch; all intermediates in LDS.
// Outputs: V (B,4,64,256) fp32 then A (B,64,64) fp32, concatenated in d_out.
// Round 3: 8-deep B-fragment register arrays (forced-live) so each k-step's
// 16 global_load_b128 issue as one clause and drain while WMMAs retire,
// instead of s_wait_loadcnt 0 before every WMMA.  Theta pass split into three
// sweeps (one per Chebyshev term) to fit the 8-deep pipeline in registers.
// ---------------------------------------------------------------------------

typedef __bf16 bf16;
typedef __attribute__((ext_vector_type(16))) __bf16 v16bf;
typedef __attribute__((ext_vector_type(8)))  __bf16 v8bf;
typedef __attribute__((ext_vector_type(8)))  float  v8f;

#define NB     1024
#define NNODE  64
#define FDIM   256
#define NHEAD  4
#define NLAYER 4

#define LDH 264   // padded leading dim (bf16) for [64][256] row-major tiles
#define LDT 72    // padded leading dim (bf16) for [*][64] K-major tiles
#define LSC 68    // padded fp32 row stride for 64x64 score/adj buffers

#define WMMA_BF16(a, b, c) \
  __builtin_amdgcn_wmma_f32_16x16x32_bf16(false, (a), false, (b), (short)0, (c), false, false)

__device__ __forceinline__ v8f vzero8() {
  v8f z;
#pragma unroll
  for (int i = 0; i < 8; ++i) z[i] = 0.0f;
  return z;
}

// A-operand fragment (16x32 bf16), source row-major with leading dim ld.
// CDNA5 ISA layout: lanes 0-15 row M=lane, K {0..7,16..23}; lanes 16-31 same
// rows, K {8..15,24..31}.  Two contiguous 16B loads per lane.
__device__ __forceinline__ v16bf frag_rowmajor(const bf16* base, int ld) {
  const int lane = threadIdx.x & 31;
  const bf16* p = base + (lane & 15) * ld + ((lane >> 4) << 3);
  v8bf lo = *(const v8bf*)(p);
  v8bf hi = *(const v8bf*)(p + 16);
  v16bf r;
#pragma unroll
  for (int i = 0; i < 8; ++i) { r[i] = lo[i]; r[i + 8] = hi[i]; }
  return r;
}

// B-operand fragment (32x16 bf16), source stored K-major: element (k, n) at
// base[n*ld + k].  CDNA5 layout: lanes 0-15 col N=lane K {0..7,16..23}; lanes
// 16-31 col N=lane-16 K {8..15,24..31}.  Two contiguous 16B loads per lane.
__device__ __forceinline__ v16bf frag_kmajor(const bf16* base, int ld) {
  const int lane = threadIdx.x & 31;
  const bf16* p = base + (lane & 15) * ld + ((lane >> 4) << 3);
  v8bf lo = *(const v8bf*)(p);
  v8bf hi = *(const v8bf*)(p + 16);
  v16bf r;
#pragma unroll
  for (int i = 0; i < 8; ++i) { r[i] = lo[i]; r[i + 8] = hi[i]; }
  return r;
}

// ---------------------------------------------------------------------------
// Weight transpose+convert: dst[m][g][f] = (bf16) src[m][f][g]  (256x256 mats)
// grid.x = nmat*256 blocks (one output row each), 256 threads.
// ---------------------------------------------------------------------------
__global__ __launch_bounds__(256) void transpose_to_bf16(const float* __restrict__ src,
                                                         bf16* __restrict__ dst) {
  const int m = blockIdx.x >> 8;
  const int g = blockIdx.x & 255;
  const int f = threadIdx.x;
  dst[((size_t)m << 16) + (g << 8) + f] = (bf16)src[((size_t)m << 16) + (f << 8) + g];
}

// ---------------------------------------------------------------------------
// Fused attention + Chebyshev GCN, one workgroup per batch.
// ---------------------------------------------------------------------------
__global__ __launch_bounds__(256) void stgc_fused(
    const float* __restrict__ x, const float* __restrict__ bq, const float* __restrict__ bk,
    const bf16* __restrict__ WqT, const bf16* __restrict__ WkT, const bf16* __restrict__ thetaT,
    float* __restrict__ Vout, float* __restrict__ Aout) {
  extern __shared__ __align__(16) char smem[];
  bf16*  xb  = (bf16*)(smem + 0);       // h  row-major  [64][LDH]    33792 B
  bf16*  hT  = (bf16*)(smem + 33792);   // h  K-major    [256][LDT]   36864 B
  bf16*  qb  = (bf16*)(smem + 70656);   // q, later Z1   [64][LDH]    33792 B
  bf16*  kb  = (bf16*)(smem + 104448);  // k, later Z2   [64][LDH]    33792 B
  bf16*  Lt  = (bf16*)(smem + 138240);  // -A (symmetric)[64][LDT]     9216 B
  bf16*  Lt2 = (bf16*)(smem + 147456);  // Lt@Lt         [64][LDT]     9216 B
  float* sc  = (float*)(smem + 156672); // scores / sym  [64][LSC]    17408 B
  float* adj = (float*)(smem + 174080); // head-mean adj [64][LSC]    17408 B
  float* deg = (float*)(smem + 191488); // deg^-1/2      [64]           256 B
  // total 191744 B dynamic LDS (fits the 320KB/WGP CDNA5 budget)

  const int b    = blockIdx.x;
  const int tid  = threadIdx.x;
  const int lane = tid & 31;
  const int wave = tid >> 5;
  const int m0   = (wave & 3) * 16;   // row-tile base for 64-row outputs
  const int nblk = (wave >> 2) * 8;   // first of 8 column tiles (256-col outputs)

  // ---- Stage A: load x_b, convert to bf16, build row-major h and K-major hT
  const float* xg = x + (size_t)b * NNODE * FDIM;
  for (int it = tid; it < NNODE * FDIM / 4; it += 256) {
    const int row = it >> 6;
    const int c4  = (it & 63) << 2;
    const float4 v = *(const float4*)(xg + row * FDIM + c4);
    const float vv[4] = {v.x, v.y, v.z, v.w};
#pragma unroll
    for (int u = 0; u < 4; ++u) {
      const bf16 bv = (bf16)vv[u];
      xb[row * LDH + c4 + u]   = bv;
      hT[(c4 + u) * LDT + row] = bv;
    }
  }
  __syncthreads();

  // ---- Stage B: q/k projections: 64x256 = x(64x256) @ W(256x256) + bias
  for (int g = 0; g < 2; ++g) {
    const bf16*  WT   = g ? WkT : WqT;
    const float* bias = g ? bk : bq;
    bf16*        dst  = g ? kb : qb;
    v8f acc[8];
#pragma unroll
    for (int j = 0; j < 8; ++j) acc[j] = vzero8();
    for (int kk = 0; kk < 8; ++kk) {
      const v16bf a = frag_rowmajor(xb + m0 * LDH + kk * 32, LDH);
      v16bf bfr[8];  // all 8 tiles live at once -> distinct VGPRs, one clause
#pragma unroll
      for (int j = 0; j < 8; ++j)
        bfr[j] = frag_kmajor(WT + (nblk + j) * 16 * 256 + kk * 32, 256);
#pragma unroll
      for (int j = 0; j < 8; ++j)
        acc[j] = WMMA_BF16(a, bfr[j], acc[j]);
    }
#pragma unroll
    for (int j = 0; j < 8; ++j) {
      const int col   = (nblk + j) * 16 + (lane & 15);
      const int rbase = m0 + ((lane >> 4) << 3);
      const float bv  = bias[col];
#pragma unroll
      for (int r = 0; r < 8; ++r)
        dst[(rbase + r) * LDH + col] = (bf16)(acc[j][r] + bv);
    }
  }
  __syncthreads();

  // ---- Stage C: per-head scores + extreme-temperature softmax -> adj mean
  const float tscale = 1.25e8f;  // TEMP / sqrt(Dh) = 1e9 / 8
  for (int hh = 0; hh < NHEAD; ++hh) {
#pragma unroll
    for (int i = 0; i < 2; ++i) {               // 16 tiles over 8 waves
      const int t  = wave * 2 + i;
      const int mt = (t >> 2) * 16, nt = (t & 3) * 16;
      v8f acc = vzero8();
#pragma unroll
      for (int kk = 0; kk < 2; ++kk) {
        const v16bf a  = frag_rowmajor(qb + mt * LDH + hh * 64 + kk * 32, LDH);
        const v16bf bb = frag_kmajor(kb + nt * LDH + hh * 64 + kk * 32, LDH); // k row-major == k^T K-major
        acc = WMMA_BF16(a, bb, acc);
      }
      const int col   = nt + (lane & 15);
      const int rbase = mt + ((lane >> 4) << 3);
#pragma unroll
      for (int r = 0; r < 8; ++r) sc[(rbase + r) * LSC + col] = acc[r];
    }
    __syncthreads();
    if (tid < 64) {  // softmax row, accumulate 1/H of it into adj
      float mx = -3.4e38f;
      for (int c = 0; c < 64; ++c) mx = fmaxf(mx, sc[tid * LSC + c]);
      float sum = 0.0f;
      for (int c = 0; c < 64; ++c) {
        const float e = __expf((sc[tid * LSC + c] - mx) * tscale);
        sc[tid * LSC + c] = e;
        sum += e;
      }
      const float inv = 0.25f / sum;
      for (int c = 0; c < 64; ++c) {
        const float v = sc[tid * LSC + c] * inv;
        if (hh == 0) adj[tid * LSC + c] = v; else adj[tid * LSC + c] += v;
      }
    }
    __syncthreads();
  }

  // ---- Symmetrize, normalize: A = D^-1/2 (adj+adj^T)/2 D^-1/2 ; Lt = -A
  for (int idx = tid; idx < 4096; idx += 256) {
    const int i = idx >> 6, j = idx & 63;
    sc[i * LSC + j] = 0.5f * (adj[i * LSC + j] + adj[j * LSC + i]);
  }
  __syncthreads();
  if (tid < 64) {
    float s = 0.0f;
    for (int c = 0; c < 64; ++c) s += sc[tid * LSC + c];
    deg[tid] = rsqrtf(s);
  }
  __syncthreads();
  float* Ab = Aout + (size_t)b * 4096;
  for (int idx = tid; idx < 4096; idx += 256) {
    const int i = idx >> 6, j = idx & 63;
    const float a = deg[i] * deg[j] * sc[i * LSC + j];
    Ab[idx] = a;                       // output A
    Lt[i * LDT + j] = (bf16)(-a);      // Lt = -A (symmetric!)
  }
  __syncthreads();

  // ---- Lt2 = Lt @ Lt  (symmetry: row-major Lt doubles as K-major B-operand)
#pragma unroll
  for (int i = 0; i < 2; ++i) {
    const int t  = wave * 2 + i;
    const int mt = (t >> 2) * 16, nt = (t & 3) * 16;
    v8f acc = vzero8();
#pragma unroll
    for (int kk = 0; kk < 2; ++kk) {
      const v16bf a  = frag_rowmajor(Lt + mt * LDT + kk * 32, LDT);
      const v16bf bb = frag_kmajor(Lt + nt * LDT + kk * 32, LDT);
      acc = WMMA_BF16(a, bb, acc);
    }
    const int col   = nt + (lane & 15);
    const int rbase = mt + ((lane >> 4) << 3);
#pragma unroll
    for (int r = 0; r < 8; ++r) Lt2[(rbase + r) * LDT + col] = (bf16)acc[r];
  }
  __syncthreads();

  // ---- Stage D: Chebyshev layers.  Z1 = Lt@h, Z2 = 2*Lt2@h - h,
  //      out = relu(Z0@th0 + Z1@th1 + Z2@th2);  h <- out
  for (int l = 0; l < NLAYER; ++l) {
    {  // Z1 and Z2 in one sweep; 8-deep LDS B-fragment array shared by both
      v8f a1[8], a2[8];
#pragma unroll
      for (int j = 0; j < 8; ++j) { a1[j] = vzero8(); a2[j] = vzero8(); }
      for (int kk = 0; kk < 2; ++kk) {
        const v16bf aL  = frag_rowmajor(Lt  + m0 * LDT + kk * 32, LDT);
        const v16bf aL2 = frag_rowmajor(Lt2 + m0 * LDT + kk * 32, LDT);
        v16bf bfr[8];
#pragma unroll
        for (int j = 0; j < 8; ++j)
          bfr[j] = frag_kmajor(hT + (nblk + j) * 16 * LDT + kk * 32, LDT);
#pragma unroll
        for (int j = 0; j < 8; ++j) {
          a1[j] = WMMA_BF16(aL,  bfr[j], a1[j]);
          a2[j] = WMMA_BF16(aL2, bfr[j], a2[j]);
        }
      }
#pragma unroll
      for (int j = 0; j < 8; ++j) {
        const int col   = (nblk + j) * 16 + (lane & 15);
        const int rbase = m0 + ((lane >> 4) << 3);
#pragma unroll
        for (int r = 0; r < 8; ++r) {
          const int row = rbase + r;
          qb[row * LDH + col] = (bf16)a1[j][r];                       // Z1
          const float h0 = (float)xb[row * LDH + col];                // Z0
          kb[row * LDH + col] = (bf16)(2.0f * a2[j][r] - h0);         // Z2
        }
      }
    }
    __syncthreads();

    // out = Z0@th0 + Z1@th1 + Z2@th2 as three sweeps so each k-step keeps an
    // 8-deep B-fragment pipeline in registers (same total loads/WMMAs).
    v8f acc[8];
#pragma unroll
    for (int j = 0; j < 8; ++j) acc[j] = vzero8();
#pragma unroll 1
    for (int c = 0; c < 3; ++c) {
      const bf16* thc = thetaT + (size_t)(l * 3 + c) * 65536;
      const bf16* src = (c == 0) ? xb : (c == 1) ? qb : kb;  // Z0 / Z1 / Z2
      for (int kk = 0; kk < 8; ++kk) {
        const v16bf a = frag_rowmajor(src + m0 * LDH + kk * 32, LDH);
        v16bf bfr[8];
#pragma unroll
        for (int j = 0; j < 8; ++j)
          bfr[j] = frag_kmajor(thc + (nblk + j) * 16 * 256 + kk * 32, 256);
#pragma unroll
        for (int j = 0; j < 8; ++j)
          acc[j] = WMMA_BF16(a, bfr[j], acc[j]);
      }
    }
    __syncthreads();  // everyone finished reading old h before overwrite

    float* Vb = Vout + (((size_t)b * NLAYER + l) * NNODE) * FDIM;
#pragma unroll
    for (int j = 0; j < 8; ++j) {
      const int col   = (nblk + j) * 16 + (lane & 15);
      const int rbase = m0 + ((lane >> 4) << 3);
#pragma unroll
      for (int r = 0; r < 8; ++r) {
        const int row = rbase + r;
        const float v = fmaxf(acc[j][r], 0.0f);
        Vb[row * FDIM + col] = v;        // output V[b][l]
        const bf16 bv = (bf16)v;
        xb[row * LDH + col]   = bv;      // next-layer h
        hT[col * LDT + row]   = bv;      // next-layer h (K-major)
      }
    }
    __syncthreads();
  }
}

extern "C" void kernel_launch(void* const* d_in, const int* in_sizes, int n_in,
                              void* d_out, int out_size, void* d_ws, size_t ws_size,
                              hipStream_t stream) {
  const float* x  = (const float*)d_in[0];
  const float* Wq = (const float*)d_in[1];
  const float* bq = (const float*)d_in[2];
  const float* Wk = (const float*)d_in[3];
  const float* bk = (const float*)d_in[4];
  const float* th = (const float*)d_in[5];

  // ws: WqT (128KB) | WkT (128KB) | thetaT (1.5MB), all bf16, pre-transposed
  bf16* WqT = (bf16*)d_ws;
  bf16* WkT = WqT + 65536;
  bf16* thT = WkT + 65536;

  transpose_to_bf16<<<dim3(256), dim3(256), 0, stream>>>(Wq, WqT);
  transpose_to_bf16<<<dim3(256), dim3(256), 0, stream>>>(Wk, WkT);
  transpose_to_bf16<<<dim3(12 * 256), dim3(256), 0, stream>>>(th, thT);

  float* Vout = (float*)d_out;
  float* Aout = Vout + (size_t)NB * NLAYER * NNODE * FDIM;

  const int lds_bytes = 191744;
  (void)hipFuncSetAttribute((const void*)stgc_fused,
                            hipFuncAttributeMaxDynamicSharedMemorySize, lds_bytes);
  stgc_fused<<<dim3(NB), dim3(256), lds_bytes, stream>>>(x, bq, bk, WqT, WkT, thT, Vout, Aout);
}
